// GroupAgent_63694365000066
// MI455X (gfx1250) — matrix operand order repaired and checked by reading, not compiled, not run on hardware
//
#include <hip/hip_runtime.h>

typedef __attribute__((ext_vector_type(16))) _Float16 v16h;
typedef __attribute__((ext_vector_type(8)))  float    v8f;

union FragA { v16h v; _Float16 h[16]; uint4 q[2]; };

__device__ __forceinline__ float sigmoidf_(float x) {
    return 1.0f / (1.0f + __expf(-x));
}

// ---------------------------------------------------------------------------
// CDNA5 async global->LDS copy (ASYNCcnt-tracked), 16B per lane.
//   lds_off: wave-relative LDS byte address (low 32 bits of flat &shared)
//   base   : wave-uniform global base (SGPR pair), gofs: per-lane byte offset
// ---------------------------------------------------------------------------
__device__ __forceinline__ void async_ld16(unsigned lds_off, const void* base,
                                           unsigned gofs) {
    asm volatile("global_load_async_to_lds_b128 %0, %1, %2"
                 :
                 : "v"(lds_off), "v"(gofs), "s"(base)
                 : "memory");
}
__device__ __forceinline__ void wait_async0() {
    asm volatile("s_wait_asynccnt 0x0" ::: "memory");
}

// Stage Bt[col0..col0+63][0..255] (f16, pre-transposed) into buf[64][264].
// 128 threads: thread t owns (col = t>>1, half-row = t&1) -> 16 x 16B units.
__device__ __forceinline__ void stage_async(_Float16 (*buf)[264],
                                            const _Float16* Wt, int col0, int t) {
    const int col = t >> 1, half = t & 1;
    const unsigned lbase = (unsigned)(size_t)&buf[col][half * 128];
    const unsigned gofs  = (unsigned)(((col0 + col) * 256 + half * 128) * 2);
    #pragma unroll
    for (int u = 0; u < 16; ++u)
        async_ld16(lbase + u * 16, Wt, gofs + u * 16);
}

// ---------------------------------------------------------------------------
// WMMA GEMM:  out = act(A @ Bt^T + bias)   [K fixed at 256]
//   A : [M,256] f16 row-major,  Bt: [Nc,256] f16 (B transposed)
//   Block: 128 thr / 4 waves -> 64x64 C tile; single async B stage.
// ---------------------------------------------------------------------------
template<int ACT>
__global__ __launch_bounds__(128)
void gemm_wmma(const _Float16* __restrict__ A, const _Float16* __restrict__ Bt,
               const float* __restrict__ bias, _Float16* __restrict__ outH,
               int M, int Nc)
{
    __shared__ __align__(16) _Float16 Bs[64][264];   // [n][k], pad 8: aligned,
                                                     // conflict-free b128 reads
    const int t    = threadIdx.x;
    const int lane = t & 31;
    const int wave = t >> 5;
    const int r0   = blockIdx.x * 64 + wave * 16;
    const int col0 = blockIdx.y * 64;
    const int mrow = r0 + (lane & 15);

    stage_async(Bs, Bt, col0, t);        // async B tile (full K)

    // A fragments per ISA 16-bit A layout: lane<16 K{0-7,16-23}, else K{8-15,24-31}
    FragA af[8];
    {
        const _Float16* ap = A + (size_t)mrow * 256 + (lane >> 4) * 8;
        #pragma unroll
        for (int kf = 0; kf < 8; ++kf) {
            af[kf].q[0] = *(const uint4*)(ap + kf * 32);
            af[kf].q[1] = *(const uint4*)(ap + kf * 32 + 16);
        }
    }

    wait_async0();
    __syncthreads();

    v8f acc[4] = {};
    const int nn = lane & 15;
    const int ko = (lane >> 4) * 8;      // B frag: low lanes K{0-7,16-23}, high K{8-15,24-31}
    #pragma unroll
    for (int kf = 0; kf < 8; ++kf) {
        #pragma unroll
        for (int ct = 0; ct < 4; ++ct) {
            FragA bf;
            bf.q[0] = *(const uint4*)&Bs[ct * 16 + nn][kf * 32 + ko];
            bf.q[1] = *(const uint4*)&Bs[ct * 16 + nn][kf * 32 + ko + 16];
            acc[ct] = __builtin_amdgcn_wmma_f32_16x16x32_f16(
                false, af[kf].v, false, bf.v, (short)0, acc[ct], false, false);
        }
    }

    #pragma unroll
    for (int ct = 0; ct < 4; ++ct) {
        const int col = col0 + ct * 16 + (lane & 15);
        const float bv = bias[col];
        #pragma unroll
        for (int i = 0; i < 8; ++i) {
            const int m = r0 + ((lane < 16) ? i : i + 8);
            float v = acc[ct][i] + bv;
            if (ACT == 1) v = fmaxf(v, 0.0f);
            if (ACT == 2) v = tanhf(v);
            outH[(size_t)m * Nc + col] = (_Float16)v;
        }
    }
}

// ---------------------------------------------------------------------------
// GRU gates (torch GRUCell, gate order r,z,n); writes hn in f32 and f16
// ---------------------------------------------------------------------------
__global__ void gru_kernel(const _Float16* __restrict__ gi,
                           const _Float16* __restrict__ gh,
                           const float* __restrict__ h,
                           float* __restrict__ hn_f,
                           _Float16* __restrict__ hn_h, int N)
{
    int idx = blockIdx.x * blockDim.x + threadIdx.x;
    if (idx >= N * 256) return;
    int n = idx >> 8, j = idx & 255;
    size_t b = (size_t)n * 768;
    float ir  = (float)gi[b + j],        hr  = (float)gh[b + j];
    float iz  = (float)gi[b + 256 + j],  hz  = (float)gh[b + 256 + j];
    float in_ = (float)gi[b + 512 + j],  hnv = (float)gh[b + 512 + j];
    float r  = sigmoidf_(ir + hr);
    float z  = sigmoidf_(iz + hz);
    float ng = tanhf(in_ + r * hnv);
    float v  = (1.0f - z) * ng + z * h[idx];
    hn_f[idx] = v;
    hn_h[idx] = (_Float16)v;
}

// q[n,a] = hb_b[a] + g[n,:]@hb_w[:,a] + hn[n,:]@hw_b(reshaped [H,A])[:,a]
__global__ void qinit_kernel(const _Float16* __restrict__ g,
                             const float* __restrict__ hn,
                             const float* __restrict__ hb_w,
                             const float* __restrict__ hb_b,
                             const float* __restrict__ hw_b,
                             float* __restrict__ q, int N)
{
    int idx = blockIdx.x * blockDim.x + threadIdx.x;
    if (idx >= N * 32) return;
    int n = idx >> 5, a = idx & 31;
    float s = hb_b[a];
    const _Float16* gr = g  + (size_t)n * 256;
    const float*    hr = hn + (size_t)n * 256;
    for (int e = 0; e < 256; ++e)    s += (float)gr[e] * hb_w[e * 32 + a];
    for (int hh = 0; hh < 256; ++hh) s += hr[hh] * hw_b[hh * 32 + a];
    q[idx] = s;
}

__global__ void cvt_kernel(const float* __restrict__ in,
                           _Float16* __restrict__ out, int n)
{
    int i = blockIdx.x * blockDim.x + threadIdx.x;
    if (i < n) out[i] = (_Float16)in[i];
}

// transpose+convert: in f32 [K,Nc] -> out f16 [Nc,K]
__global__ void cvt_t_kernel(const float* __restrict__ in,
                             _Float16* __restrict__ out, int K, int Nc)
{
    int idx = blockIdx.x * blockDim.x + threadIdx.x;
    if (idx >= K * Nc) return;
    int k = idx % K, n = idx / K;
    out[idx] = (_Float16)in[(size_t)k * Nc + n];
}

// ---------------------------------------------------------------------------
// Fused hyper GEMM + einsum: q[n,a] += sum_h hn[n,h] * (g[n,:]@hw_w[:,h*32+a])
// Block = 64 rows x all 8192 cols (128 chunks). Double-buffered async B tiles
// overlap the copy of chunk c+1 with the 32 WMMAs of chunk c. Each 16-col tile
// maps to one h; single writer per q element -> deterministic, no atomics.
// ---------------------------------------------------------------------------
__global__ __launch_bounds__(128)
void hyper_kernel(const _Float16* __restrict__ G,    // [N,256] f16
                  const _Float16* __restrict__ Wt,   // [8192,256] f16 (hw_w^T)
                  const float* __restrict__ hn,      // [N,256] f32
                  float* __restrict__ q)             // [N,32] f32, pre-init
{
    __shared__ __align__(16) _Float16 Bs[2][64][264];
    __shared__ __align__(16) float    Hs[64][256];

    const int t    = threadIdx.x;
    const int lane = t & 31;
    const int wave = t >> 5;
    const int r0   = blockIdx.x * 64;
    const int mrow = r0 + wave * 16 + (lane & 15);

    // async preload hn tile (64x256 f32 = 4096 x 16B units, 32 per thread)
    {
        const unsigned hbase = (unsigned)(size_t)&Hs[0][0];
        const float* hsrc = hn + (size_t)r0 * 256;
        #pragma unroll
        for (int u = 0; u < 32; ++u) {
            const unsigned ofs = (unsigned)((u * 128 + t) * 16);
            async_ld16(hbase + ofs, hsrc, ofs);
        }
    }

    // this wave's 8 A-fragments of g (K = 0..255), kept in registers
    FragA af[8];
    {
        const _Float16* ap = G + (size_t)mrow * 256 + (lane >> 4) * 8;
        #pragma unroll
        for (int kf = 0; kf < 8; ++kf) {
            af[kf].q[0] = *(const uint4*)(ap + kf * 32);
            af[kf].q[1] = *(const uint4*)(ap + kf * 32 + 16);
        }
    }

    stage_async(Bs[0], Wt, 0, t);        // chunk 0 -> buffer 0

    v8f qacc[2] = {};
    const int nn = lane & 15;
    const int ko = (lane >> 4) * 8;

    for (int c = 0; c < 128; ++c) {
        const int cur = c & 1;
        wait_async0();                    // chunk c (and Hs on c==0) landed
        __syncthreads();                  // visible to all waves
        if (c + 1 < 128)
            stage_async(Bs[cur ^ 1], Wt, (c + 1) * 64, t);   // overlap copy

        const int h0 = c * 2;             // chunk covers h0, h0+1
        #pragma unroll
        for (int ct = 0; ct < 4; ++ct) {  // (h = h0 + (ct>>1), a-half = ct&1)
            v8f acc = {};
            #pragma unroll
            for (int kf = 0; kf < 8; ++kf) {
                FragA bf;
                bf.q[0] = *(const uint4*)&Bs[cur][ct * 16 + nn][kf * 32 + ko];
                bf.q[1] = *(const uint4*)&Bs[cur][ct * 16 + nn][kf * 32 + ko + 16];
                acc = __builtin_amdgcn_wmma_f32_16x16x32_f16(
                    false, af[kf].v, false, bf.v, (short)0, acc, false, false);
            }
            const int hsel = h0 + (ct >> 1);
            const int dst  = ct & 1;
            #pragma unroll
            for (int i = 0; i < 8; ++i) {
                const int mr = wave * 16 + ((lane < 16) ? i : i + 8);
                qacc[dst][i] += acc[i] * Hs[mr][hsel];
            }
        }
    }

    // single-writer accumulate into q (bias terms from qinit)
    #pragma unroll
    for (int dst = 0; dst < 2; ++dst) {
        const int a = dst * 16 + (lane & 15);
        #pragma unroll
        for (int i = 0; i < 8; ++i) {
            const int m = r0 + wave * 16 + ((lane < 16) ? i : i + 8);
            q[(size_t)m * 32 + a] += qacc[dst][i];
        }
    }
}

// ---------------------------------------------------------------------------
extern "C" void kernel_launch(void* const* d_in, const int* in_sizes, int n_in,
                              void* d_out, int out_size, void* d_ws, size_t ws_size,
                              hipStream_t stream)
{
    (void)in_sizes; (void)n_in; (void)out_size; (void)ws_size;

    const float* x     = (const float*)d_in[0];
    const float* h     = (const float*)d_in[1];
    const float* fc1_w = (const float*)d_in[2];
    const float* fc1_b = (const float*)d_in[3];
    const float* w_ih  = (const float*)d_in[4];
    const float* b_ih  = (const float*)d_in[5];
    const float* w_hh  = (const float*)d_in[6];
    const float* b_hh  = (const float*)d_in[7];
    const float* hg_w1 = (const float*)d_in[8];
    const float* hg_b1 = (const float*)d_in[9];
    const float* hg_w2 = (const float*)d_in[10];
    const float* hg_b2 = (const float*)d_in[11];
    const float* hb_w  = (const float*)d_in[12];
    const float* hb_b  = (const float*)d_in[13];
    const float* hw_w  = (const float*)d_in[14];
    const float* hw_b  = (const float*)d_in[15];
    float* q = (float*)d_out;

    const int N = 16384;
    char* ws = (char*)d_ws;
    size_t off = 0;
    auto take = [&](size_t bytes) {
        void* p = (void*)(ws + off);
        off += (bytes + 255) & ~(size_t)255;
        return p;
    };
    _Float16* x_h   = (_Float16*)take((size_t)N * 256 * 2);
    _Float16* h_h   = (_Float16*)take((size_t)N * 256 * 2);
    _Float16* fc1_t = (_Float16*)take((size_t)256 * 256 * 2);
    _Float16* wih_t = (_Float16*)take((size_t)256 * 768 * 2);
    _Float16* whh_t = (_Float16*)take((size_t)256 * 768 * 2);
    _Float16* hg1_t = (_Float16*)take((size_t)256 * 256 * 2);
    _Float16* hg2_t = (_Float16*)take((size_t)256 * 256 * 2);
    _Float16* hw_t  = (_Float16*)take((size_t)256 * 8192 * 2);
    _Float16* x1_h  = (_Float16*)take((size_t)N * 256 * 2);
    _Float16* gi_h  = (_Float16*)take((size_t)N * 768 * 2);
    _Float16* gh_h  = (_Float16*)take((size_t)N * 768 * 2);
    float*    hn_f  = (float*)   take((size_t)N * 256 * 4);
    _Float16* hn_h  = (_Float16*)take((size_t)N * 256 * 2);
    _Float16* t_h   = (_Float16*)take((size_t)N * 256 * 2);
    _Float16* g_h   = (_Float16*)take((size_t)N * 256 * 2);

    dim3 blk(128);

    // precision conversions + weight transposes (all L2-resident, cheap)
    cvt_kernel<<<(N * 256 + 255) / 256, 256, 0, stream>>>(x, x_h, N * 256);
    cvt_kernel<<<(N * 256 + 255) / 256, 256, 0, stream>>>(h, h_h, N * 256);
    cvt_t_kernel<<<(256 * 256 + 255) / 256, 256, 0, stream>>>(fc1_w, fc1_t, 256, 256);
    cvt_t_kernel<<<(256 * 768 + 255) / 256, 256, 0, stream>>>(w_ih, wih_t, 256, 768);
    cvt_t_kernel<<<(256 * 768 + 255) / 256, 256, 0, stream>>>(w_hh, whh_t, 256, 768);
    cvt_t_kernel<<<(256 * 256 + 255) / 256, 256, 0, stream>>>(hg_w1, hg1_t, 256, 256);
    cvt_t_kernel<<<(256 * 256 + 255) / 256, 256, 0, stream>>>(hg_w2, hg2_t, 256, 256);
    cvt_t_kernel<<<(256 * 8192 + 255) / 256, 256, 0, stream>>>(hw_w, hw_t, 256, 8192);

    // x1 = relu(x @ fc1_w + fc1_b)
    gemm_wmma<1><<<dim3(N / 64, 256 / 64), blk, 0, stream>>>(x_h, fc1_t, fc1_b, x1_h, N, 256);
    // gi = x1 @ w_ih + b_ih ; gh = h @ w_hh + b_hh
    gemm_wmma<0><<<dim3(N / 64, 768 / 64), blk, 0, stream>>>(x1_h, wih_t, b_ih, gi_h, N, 768);
    gemm_wmma<0><<<dim3(N / 64, 768 / 64), blk, 0, stream>>>(h_h, whh_t, b_hh, gh_h, N, 768);
    // hn = GRU(gi, gh, h)
    gru_kernel<<<(N * 256 + 255) / 256, 256, 0, stream>>>(gi_h, gh_h, h, hn_f, hn_h, N);
    // t = relu(hn @ hg_w1 + b1) ; g = tanh(t @ hg_w2 + b2)
    gemm_wmma<1><<<dim3(N / 64, 256 / 64), blk, 0, stream>>>(hn_h, hg1_t, hg_b1, t_h, N, 256);
    gemm_wmma<2><<<dim3(N / 64, 256 / 64), blk, 0, stream>>>(t_h, hg2_t, hg_b2, g_h, N, 256);
    // q = hb_b + g@hb_w + hn@hw_b(reshaped)
    qinit_kernel<<<(N * 32 + 255) / 256, 256, 0, stream>>>(g_h, hn_f, hb_w, hb_b, hw_b, q, N);
    // q += einsum(hn, reshape(g @ hw_w))  -- fused, async double-buffered
    hyper_kernel<<<N / 64, blk, 0, stream>>>(g_h, hw_t, hn_f, q);
}
